// RAGSequentialRec_6365141533105
// MI455X (gfx1250) — matrix operand order repaired and checked by reading, not compiled, not run on hardware
//
#include <hip/hip_runtime.h>
#include <hip/hip_bf16.h>
#include <math.h>

#define N_ITEMS 100000
#define HIDDEN  512
#define BATCH   1024
#define SEQ_LEN 200
#define TOP_K   50

typedef __attribute__((ext_vector_type(16))) __bf16 v16bf;
typedef __attribute__((ext_vector_type(8)))  __bf16 v8bf;
typedef __attribute__((ext_vector_type(8)))  float  v8f;
typedef __attribute__((ext_vector_type(4)))  unsigned v4u;
typedef __attribute__((ext_vector_type(8)))  int    v8i;
typedef __attribute__((ext_vector_type(4)))  int    v4i;

#if defined(__HIP_DEVICE_COMPILE__) && defined(__gfx1250__)
#define ON_GFX1250 1
#else
#define ON_GFX1250 0
#endif

// ---------------------------------------------------------------------------
// Kernel 1: masked mean-pool of sequence embeddings + mean of retrieved embs
// One block per batch row; thread t covers columns t and t+256.
// ---------------------------------------------------------------------------
__global__ __launch_bounds__(256) void pool_kernel(
    const int* __restrict__ seq_ids, const int* __restrict__ ret_idx,
    const float* __restrict__ emb, float* __restrict__ pooled,
    float* __restrict__ retrieved) {
  const int b = blockIdx.x;
  const int t = threadIdx.x;
  const int h0 = t, h1 = t + 256;
  float a0 = 0.f, a1 = 0.f;
  int cnt = 0;
  for (int s = 0; s < SEQ_LEN; ++s) {
    const int id = seq_ids[b * SEQ_LEN + s];
    if (id != 0) {
      const float* e = emb + (size_t)(id - 1) * HIDDEN;
      a0 += e[h0];
      a1 += e[h1];
      ++cnt;
    }
  }
  const float inv = 1.0f / (float)(cnt > 0 ? cnt : 1);
  pooled[(size_t)b * HIDDEN + h0] = a0 * inv;
  pooled[(size_t)b * HIDDEN + h1] = a1 * inv;

  float r0 = 0.f, r1 = 0.f;
  for (int k = 0; k < TOP_K; ++k) {
    const float* e = emb + (size_t)ret_idx[b * TOP_K + k] * HIDDEN;
    r0 += e[h0];
    r1 += e[h1];
  }
  retrieved[(size_t)b * HIDDEN + h0] = r0 * (1.0f / TOP_K);
  retrieved[(size_t)b * HIDDEN + h1] = r1 * (1.0f / TOP_K);
}

// ---------------------------------------------------------------------------
// Kernel 2: user_rep = pooled @ W_rec.T + b_rec   (f32, exact)
// Block handles 8 m-rows; thread t computes columns t and t+256.
// ---------------------------------------------------------------------------
__global__ __launch_bounds__(256) void userrep_kernel(
    const float* __restrict__ pooled, const float* __restrict__ W_rec,
    const float* __restrict__ b_rec, float* __restrict__ user_rep) {
  const int m0 = blockIdx.x * 8;
  const int t = threadIdx.x;
  const int n0 = t, n1 = t + 256;
  float acc[8][2] = {};
  for (int k = 0; k < HIDDEN; ++k) {
    const float w0 = W_rec[(size_t)n0 * HIDDEN + k];
    const float w1 = W_rec[(size_t)n1 * HIDDEN + k];
#pragma unroll
    for (int i = 0; i < 8; ++i) {
      const float a = pooled[(size_t)(m0 + i) * HIDDEN + k];
      acc[i][0] = fmaf(a, w0, acc[i][0]);
      acc[i][1] = fmaf(a, w1, acc[i][1]);
    }
  }
  const float bb0 = b_rec[n0], bb1 = b_rec[n1];
#pragma unroll
  for (int i = 0; i < 8; ++i) {
    user_rep[(size_t)(m0 + i) * HIDDEN + n0] = acc[i][0] + bb0;
    user_rep[(size_t)(m0 + i) * HIDDEN + n1] = acc[i][1] + bb1;
  }
}

// ---------------------------------------------------------------------------
// Kernel 3: gate = sigmoid([user_rep, retrieved] @ gate_W.T + gate_b)
//           fused = gate*user_rep + (1-gate)*retrieved  -> bf16 (GEMM A)
// ---------------------------------------------------------------------------
__global__ __launch_bounds__(256) void fuse_kernel(
    const float* __restrict__ user_rep, const float* __restrict__ retrieved,
    const float* __restrict__ gate_W, const float* __restrict__ gate_b,
    __bf16* __restrict__ fusedb) {
  const int m0 = blockIdx.x * 8;
  const int t = threadIdx.x;
  const int n0 = t, n1 = t + 256;
  float acc[8][2] = {};
  for (int k = 0; k < HIDDEN; ++k) {
    const float w0 = gate_W[(size_t)n0 * (2 * HIDDEN) + k];
    const float w1 = gate_W[(size_t)n1 * (2 * HIDDEN) + k];
#pragma unroll
    for (int i = 0; i < 8; ++i) {
      const float a = user_rep[(size_t)(m0 + i) * HIDDEN + k];
      acc[i][0] = fmaf(a, w0, acc[i][0]);
      acc[i][1] = fmaf(a, w1, acc[i][1]);
    }
  }
  for (int k = 0; k < HIDDEN; ++k) {
    const float w0 = gate_W[(size_t)n0 * (2 * HIDDEN) + HIDDEN + k];
    const float w1 = gate_W[(size_t)n1 * (2 * HIDDEN) + HIDDEN + k];
#pragma unroll
    for (int i = 0; i < 8; ++i) {
      const float a = retrieved[(size_t)(m0 + i) * HIDDEN + k];
      acc[i][0] = fmaf(a, w0, acc[i][0]);
      acc[i][1] = fmaf(a, w1, acc[i][1]);
    }
  }
  const float gb0 = gate_b[n0], gb1 = gate_b[n1];
#pragma unroll
  for (int i = 0; i < 8; ++i) {
    const size_t row = (size_t)(m0 + i) * HIDDEN;
    const float g0 = 1.0f / (1.0f + expf(-(acc[i][0] + gb0)));
    const float g1 = 1.0f / (1.0f + expf(-(acc[i][1] + gb1)));
    const float f0 = g0 * user_rep[row + n0] + (1.0f - g0) * retrieved[row + n0];
    const float f1 = g1 * user_rep[row + n1] + (1.0f - g1) * retrieved[row + n1];
    fusedb[row + n0] = (__bf16)f0;
    fusedb[row + n1] = (__bf16)f1;
  }
}

// ---------------------------------------------------------------------------
// Kernel 4: logits = fused @ proj_W.T + proj_b  via v_wmma_f32_16x16x32_bf16
// Block tile 64(M) x 32(N), K chunk 128 staged in LDS.
// A (bf16) chunk pulled by the Tensor Data Mover (pad_enable reproduces the
// 136-half LDS row stride); B (f32 proj_W) converted to bf16 on the fly.
// 8 wave32s, each owning one 16x16 f32 accumulator.
// ---------------------------------------------------------------------------
#define LDT 136  // halves per LDS row: 128 data + 8 pad -> conflict-free

__global__ __launch_bounds__(256) void gemm_kernel(
    const __bf16* __restrict__ fusedb, const float* __restrict__ proj_W,
    const float* __restrict__ proj_b, float* __restrict__ out) {
  __shared__ __attribute__((aligned(16))) __bf16 As[64 * LDT];
  __shared__ __attribute__((aligned(16))) __bf16 Bs[32 * LDT];

  const int t = threadIdx.x;
  const int lane = t & 31;
  const int w = t >> 5;
  const int m0 = (w & 3) * 16;   // wave's M tile inside block
  const int n0 = (w >> 2) * 16;  // wave's N tile inside block
  const int m0g = blockIdx.y * 64;
  const int n0g = blockIdx.x * 32;

  v8f c = {};

  const unsigned ldsA = (unsigned)(unsigned long long)(size_t)&As[0];
  const unsigned long long gA0 =
      (unsigned long long)(size_t)fusedb + (unsigned long long)m0g * HIDDEN * 2ull;

  for (int kc = 0; kc < 4; ++kc) {
    const int kbase = kc * 128;

    // --- stage B: proj_W[n0g..n0g+31][kbase..kbase+127] f32 -> bf16 LDS ---
    {
      const int r = t >> 3;          // 0..31
      const int c0 = (t & 7) * 16;   // 0..112
      const float* src = proj_W + (size_t)(n0g + r) * HIDDEN + kbase + c0;
      __bf16* dst = &Bs[r * LDT + c0];
#pragma unroll
      for (int i = 0; i < 16; i += 4) {
        const float4 f = *(const float4*)(src + i);
        dst[i + 0] = (__bf16)f.x;
        dst[i + 1] = (__bf16)f.y;
        dst[i + 2] = (__bf16)f.z;
        dst[i + 3] = (__bf16)f.w;
      }
      if (kc < 3) __builtin_prefetch(src + 128, 0, 1);
    }

    // --- stage A: fused_bf16 tile 64 rows x 128 halves via Tensor Data Mover ---
#if ON_GFX1250
    if (w == 0) {
      const unsigned long long ga = gA0 + (unsigned long long)kbase * 2ull;
      // D# group 0: count=1 | lds_addr | global_addr | type=2
      const v4u g0 = {1u,
                      ldsA,
                      (unsigned)(ga & 0xFFFFFFFFu),
                      (unsigned)((ga >> 32) & 0x1FFFFFFu) | (2u << 30)};
      // D# group 1: data_size=2B, pad every 64 dwords by 4 dwords (-> 136-half
      // row stride), tensor 512x1024, tile 128(K) x 64(M), dim0 stride 512.
      const v8i g1 = {(int)((1u << 16) | (1u << 20) | (5u << 22) | (3u << 25)),
                      (int)((unsigned)HIDDEN << 16),   // tensor_dim0 = 512
                      (int)((unsigned)BATCH << 16),    // tensor_dim1 = 1024
                      (int)(128u << 16),               // tile_dim0 = 128
                      64,                              // tile_dim1 = 64
                      HIDDEN,                          // tensor_dim0_stride
                      0, 0};
      const v4i g2 = {1, 1, 0, 0};  // tensor_dim2/3 = 1, tile_dim3 unused
      const v4i g3 = {0, 0, 0, 0};
#if __has_builtin(__builtin_amdgcn_tensor_load_to_lds) && (__clang_major__ <= 22)
      __builtin_amdgcn_tensor_load_to_lds(g0, g1, g2, g3, 0);
#else
      asm volatile("tensor_load_to_lds %0, %1, %2, %3"
                   :
                   : "s"(g0), "s"(g1), "s"(g2), "s"(g3)
                   : "memory");
#endif
#if __has_builtin(__builtin_amdgcn_s_wait_tensorcnt)
      __builtin_amdgcn_s_wait_tensorcnt(0);
#else
      asm volatile("s_wait_tensorcnt 0x0" ::: "memory");
#endif
    }
#else
    {
      const int r = t >> 2;          // 0..63
      const int c0 = (t & 3) * 32;   // 0..96
      const __bf16* src = fusedb + (size_t)(m0g + r) * HIDDEN + kbase + c0;
      __bf16* dst = &As[r * LDT + c0];
#pragma unroll
      for (int i = 0; i < 32; i += 8)
        *(v8bf*)(dst + i) = *(const v8bf*)(src + i);
    }
#endif
    __syncthreads();

    // --- compute: 4 x v_wmma_f32_16x16x32_bf16 over this K chunk ---
    const int offA = (lane < 16) ? 0 : 8;
    const int offB = (lane < 16) ? 0 : 8;
    const __bf16* pa = &As[(m0 + (lane & 15)) * LDT];
    const __bf16* pb = &Bs[(n0 + (lane & 15)) * LDT];
#pragma unroll
    for (int kk = 0; kk < 128; kk += 32) {
      const v8bf a0 = *(const v8bf*)(pa + kk + offA);
      const v8bf a1 = *(const v8bf*)(pa + kk + 16 + offA);
      const v8bf b0 = *(const v8bf*)(pb + kk + offB);
      const v8bf b1 = *(const v8bf*)(pb + kk + 16 + offB);
      v16bf a, b;
#pragma unroll
      for (int i = 0; i < 8; ++i) {
        a[i] = a0[i];
        a[i + 8] = a1[i];
        b[i] = b0[i];
        b[i + 8] = b1[i];
      }
      c = __builtin_amdgcn_wmma_f32_16x16x32_bf16(
          /*neg_a=*/false, a, /*neg_b=*/false, b,
          /*c_mod=*/(short)0, c, /*reuse_a=*/false, /*reuse_b=*/false);
    }
    __syncthreads();
  }

  // --- epilogue: add proj_b, scatter 16x16 f32 tile ---
  const int n = n0g + n0 + (lane & 15);
  const float pbv = proj_b[n];
  const int mrow = m0g + m0 + ((lane < 16) ? 0 : 8);
#pragma unroll
  for (int v = 0; v < 8; ++v)
    out[(size_t)(mrow + v) * N_ITEMS + n] = c[v] + pbv;
}

// ---------------------------------------------------------------------------
extern "C" void kernel_launch(void* const* d_in, const int* in_sizes, int n_in,
                              void* d_out, int out_size, void* d_ws, size_t ws_size,
                              hipStream_t stream) {
  const int*   seq    = (const int*)d_in[0];
  const int*   ridx   = (const int*)d_in[1];
  const float* emb    = (const float*)d_in[2];
  const float* W_rec  = (const float*)d_in[3];
  const float* b_rec  = (const float*)d_in[4];
  const float* gate_W = (const float*)d_in[5];
  const float* gate_b = (const float*)d_in[6];
  const float* proj_W = (const float*)d_in[7];
  const float* proj_b = (const float*)d_in[8];
  float* out = (float*)d_out;

  float* pooled    = (float*)d_ws;                       // [1024*512] f32
  float* retrieved = pooled + BATCH * HIDDEN;            // [1024*512] f32
  float* user_rep  = retrieved + BATCH * HIDDEN;         // [1024*512] f32
  __bf16* fusedb   = (__bf16*)(user_rep + BATCH * HIDDEN);  // [1024*512] bf16

  pool_kernel<<<BATCH, 256, 0, stream>>>(seq, ridx, emb, pooled, retrieved);
  userrep_kernel<<<BATCH / 8, 256, 0, stream>>>(pooled, W_rec, b_rec, user_rep);
  fuse_kernel<<<BATCH / 8, 256, 0, stream>>>(user_rep, retrieved, gate_W, gate_b, fusedb);
  gemm_kernel<<<dim3(N_ITEMS / 32, BATCH / 64), 256, 0, stream>>>(fusedb, proj_W, proj_b, out);
}
